// FalconAttentionSparse_45165876084771
// MI455X (gfx1250) — matrix operand order repaired and checked by reading, not compiled, not run on hardware
//
#include <hip/hip_runtime.h>
#include <stdint.h>

#define S_LEN 2048
#define HDIM  4544
#define NQKV  4672
#define NHEADS 71
#define HD    64
#define NSEL  (S_LEN - 128)   // 1920 candidates for top-k
#define HEAVY 128

typedef __attribute__((ext_vector_type(16))) __bf16 v16bf;
typedef __attribute__((ext_vector_type(8)))  float  v8f;
typedef unsigned int v4u __attribute__((ext_vector_type(4)));
typedef int v8i_ __attribute__((ext_vector_type(8)));
typedef int v4i_ __attribute__((ext_vector_type(4)));

union BF16x16 {
  uint4 u[2];
  unsigned short s[16];
  v16bf v;
};

static __device__ __forceinline__ unsigned short f2bfu(float f) {
  unsigned int u = __float_as_uint(f);
  u += 0x7FFFu + ((u >> 16) & 1u);   // round-to-nearest-even
  return (unsigned short)(u >> 16);
}

static __device__ __forceinline__ v8f wmma_bf16(v16bf a, v16bf b, v8f c) {
  return __builtin_amdgcn_wmma_f32_16x16x32_bf16(false, a, false, b, (short)0, c, false, false);
}

static __device__ __forceinline__ v8f v8f_zero() {
  v8f z;
  for (int i = 0; i < 8; ++i) z[i] = 0.0f;
  return z;
}

static __device__ __forceinline__ unsigned lds_off(const void* p) {
  return (unsigned)(unsigned long long)(uintptr_t)p;  // low 32b of LDS aperture addr
}

// Tensor Data Mover: 2D tile load (bf16 elements) global -> LDS.
// Descriptor bit layout per CDNA5 ISA 8.3/8.4 (D# group0 / group1).
// This toolchain exposes the 6-arg builtin:
//   (uint32x4 g0, int32x8 g1, int32x4 g2, int32x4 g3, int32x8 pad, i32 cpol)
static __device__ __forceinline__ void tdm_load_2d_bf16(
    unsigned lds_addr, const void* gptr,
    unsigned tile0, unsigned tile1,
    unsigned tdim0, unsigned tdim1, unsigned stride0_elems) {
  unsigned long long ga = (unsigned long long)(uintptr_t)gptr;
  v4u g0;
  g0[0] = 1u;                                              // count=1 (user D#)
  g0[1] = lds_addr;                                        // lds_addr [63:32]
  g0[2] = (unsigned)(ga & 0xFFFFFFFFu);                    // global_addr[31:0]
  g0[3] = (unsigned)((ga >> 32) & 0x01FFFFFFu) | 0x80000000u; // addr[56:32] | type=2
  v8i_ g1;
  g1[0] = (int)(1u << 16);                                 // data_size=1 (2 bytes)
  g1[1] = (int)((tdim0 & 0xFFFFu) << 16);                  // tensor_dim0[15:0]
  g1[2] = (int)(((tdim0 >> 16) & 0xFFFFu) | ((tdim1 & 0xFFFFu) << 16));
  g1[3] = (int)(((tdim1 >> 16) & 0xFFFFu) | ((tile0 & 0xFFFFu) << 16)); // tile_dim0
  g1[4] = (int)(tile1 & 0xFFFFu);                          // tile_dim1 (tile_dim2=0)
  g1[5] = (int)stride0_elems;                              // tensor_dim0_stride[31:0]
  g1[6] = 0;                                               // stride0[47:32], stride1 lo
  g1[7] = 0;
  v4i_ z4;
  z4[0] = 0; z4[1] = 0; z4[2] = 0; z4[3] = 0;
  v8i_ z8;
  for (int i = 0; i < 8; ++i) z8[i] = 0;
  __builtin_amdgcn_tensor_load_to_lds(g0, g1, z4, z4, z8, 0);
}

// ---------------------------------------------------------------- converts
__global__ void f32_to_bf16_kernel(const float* __restrict__ in,
                                   unsigned short* __restrict__ out, int n) {
  int i = blockIdx.x * blockDim.x + threadIdx.x;
  if (i < n) out[i] = f2bfu(in[i]);
}

__global__ void zero_f32_kernel(float* __restrict__ p, int n) {
  int i = blockIdx.x * blockDim.x + threadIdx.x;
  if (i < n) p[i] = 0.0f;
}

// ------------------------------------------------------ bf16 WMMA GEMM
// C[M,N] f32 = A[M,K] bf16 @ B[K,N] bf16.  Block tile 128x64, K step 32.
// A tile staged by the Tensor Data Mover (double-buffered, wave 0 issues,
// TENSORcnt synchronized); B tile staged manually with an LDS transpose.
// 8 waves arranged 4(M) x 2(N); each wave owns 2x2 WMMA accumulators.
template <int N, int K>
__global__ __launch_bounds__(256) void gemm_bf16_kernel(
    const unsigned short* __restrict__ A, const unsigned short* __restrict__ B,
    float* __restrict__ C) {
  __shared__ __align__(16) unsigned short As[2][128][32];  // [buf][m][k] 16KB
  __shared__ __align__(16) unsigned short Bs[2][64][32];   // [buf][n][k] 8KB

  const int tid  = threadIdx.x;
  const int lane = tid & 31;
  const int wave = tid >> 5;
  const int wm   = wave >> 1;      // 0..3
  const int wn   = wave & 1;       // 0..1
  const int r    = lane & 15;
  const int hi   = lane >> 4;
  const int bm   = blockIdx.y * 128;
  const int bn   = blockIdx.x * 64;

  v8f acc[2][2];
  for (int a = 0; a < 2; ++a)
    for (int b = 0; b < 2; ++b) acc[a][b] = v8f_zero();

  const int bk  = tid >> 3;             // 0..31 (k row)
  const int bn8 = (tid & 7) << 3;       // 0..56 (n chunk)
  const unsigned short* Abase = A + (size_t)bm * K;

  // ---- prologue: stage k-slab 0
  if (wave == 0)
    tdm_load_2d_bf16(lds_off(&As[0][0][0]), Abase, 32, 128, K, 128, K);
  {
    union { uint4 q; unsigned short s[8]; } t;
    t.q = *(const uint4*)(B + (size_t)bk * N + bn + bn8);
    for (int j = 0; j < 8; ++j) Bs[0][bn8 + j][bk] = t.s[j];
  }
  if (wave == 0) __builtin_amdgcn_s_wait_tensorcnt(0);
  __syncthreads();

  const int nk = K / 32;
  for (int kt = 0; kt < nk; ++kt) {
    const int cur = kt & 1;
    const int nxt = cur ^ 1;
    const int k0n = (kt + 1) * 32;
    const bool have_next = (kt + 1 < nk);

    union { uint4 q; unsigned short s[8]; } tB;
    if (have_next) {
      if (wave == 0)                       // DMA next A slab while we compute
        tdm_load_2d_bf16(lds_off(&As[nxt][0][0]), Abase + k0n, 32, 128, K, 128, K);
      tB.q = *(const uint4*)(B + (size_t)(k0n + bk) * N + bn + bn8);
      __builtin_prefetch(B + (size_t)(k0n + 32 + bk) * N + bn + bn8, 0, 1);
    }

    // ---- fragments per ISA layout, compute on current buffers
    v16bf af[2], bfr[2];
    for (int sm = 0; sm < 2; ++sm) {          // A: row=r, K = {hi*8+j, 16+hi*8+j}
      int row = wm * 32 + sm * 16 + r;
      BF16x16 u;
      u.u[0] = *(const uint4*)&As[cur][row][hi * 8];
      u.u[1] = *(const uint4*)&As[cur][row][16 + hi * 8];
      af[sm] = u.v;
    }
    for (int sn = 0; sn < 2; ++sn) {          // B: col=r, K = {hi*8+j, 16+hi*8+j}
      int col = wn * 32 + sn * 16 + r;
      BF16x16 u;
      u.u[0] = *(const uint4*)&Bs[cur][col][hi * 8];
      u.u[1] = *(const uint4*)&Bs[cur][col][16 + hi * 8];
      bfr[sn] = u.v;
    }
    for (int sm = 0; sm < 2; ++sm)
      for (int sn = 0; sn < 2; ++sn)
        acc[sm][sn] = wmma_bf16(af[sm], bfr[sn], acc[sm][sn]);

    if (have_next) {
      for (int j = 0; j < 8; ++j) Bs[nxt][bn8 + j][bk] = tB.s[j];
      if (wave == 0) __builtin_amdgcn_s_wait_tensorcnt(0);
    }
    __syncthreads();
  }

  // C layout: VGPR i, lane (r,hi) -> row = i + 8*hi, col = r
  for (int sm = 0; sm < 2; ++sm)
    for (int sn = 0; sn < 2; ++sn) {
      float* cp = C + (size_t)(bm + wm * 32 + sm * 16 + hi * 8) * N
                    + bn + wn * 32 + sn * 16 + r;
      for (int i = 0; i < 8; ++i) cp[(size_t)i * N] = acc[sm][sn][i];
    }
}

// ---------------------------------------------------------------- RoPE
// fused[s][4672] -> Q bf16 [h][s][64], K bf16 [s][64], V bf16 [s][64]
__global__ void rope_kernel(const float* __restrict__ fused,
                            unsigned short* __restrict__ Q,
                            unsigned short* __restrict__ Kb,
                            unsigned short* __restrict__ Vb) {
  int idx = blockIdx.x * blockDim.x + threadIdx.x;   // S*73*64 total
  if (idx >= S_LEN * 73 * HD) return;
  int d    = idx & 63;
  int head = (idx >> 6) % 73;
  int s    = idx / (73 * HD);
  size_t base = (size_t)s * NQKV + head * HD;
  float x = fused[base + d];
  float y;
  if (head < 72) {
    int dm = d & 31;
    float inv = __powf(10000.0f, -(float)dm * (1.0f / 32.0f));
    float ang = (float)s * inv;
    float c  = __cosf(ang);
    float sn = __sinf(ang);
    float other = fused[base + (d ^ 32)];
    float rot = (d < 32) ? -other : other;
    y = x * c + rot * sn;
  } else {
    y = x;
  }
  if (head < NHEADS)       Q[((size_t)head * S_LEN + s) * HD + d] = f2bfu(y);
  else if (head == NHEADS) Kb[(size_t)s * HD + d] = f2bfu(y);
  else                     Vb[(size_t)s * HD + d] = f2bfu(y);
}

// ------------------------------------------------------ attention (MQA)
// grid (32, 71), block 128 = 4 waves; each wave handles one 16-query tile of
// one head.  Two-pass online softmax, WMMA for QK^T and PV, column sums of
// probs accumulated atomically into ssum[2048].
__global__ __launch_bounds__(128) void attn_kernel(
    const unsigned short* __restrict__ Q, const unsigned short* __restrict__ Kb,
    const unsigned short* __restrict__ Vb, unsigned short* __restrict__ merged,
    float* __restrict__ ssum) {
  __shared__ __align__(16) float pl[4][16][32];   // per-wave P transpose buffer

  const int lane  = threadIdx.x & 31;
  const int wave  = threadIdx.x >> 5;
  const int h     = blockIdx.y;
  const int qbase = (blockIdx.x * 4 + wave) * 16;
  const int r     = lane & 15;
  const int hi    = lane >> 4;

  // Q A-fragments: row = qbase + r, K(d) = f*32 + hi*16 + j
  v16bf qf[2];
  {
    const unsigned short* qrow = Q + ((size_t)h * S_LEN + qbase + r) * HD;
    for (int f = 0; f < 2; ++f) {
      BF16x16 u;
      u.u[0] = *(const uint4*)(qrow + f * 32 + hi * 16);
      u.u[1] = *(const uint4*)(qrow + f * 32 + hi * 16 + 8);
      qf[f] = u.v;
    }
  }

  float m[8], l[8];
  for (int i = 0; i < 8; ++i) { m[i] = -1e30f; l[i] = 0.0f; }
  const int kend = qbase + 16;

  // -------- pass 1: row max / denominator (online)
  for (int n0 = 0; n0 < kend; n0 += 16) {
    const unsigned short* krow = Kb + (size_t)(n0 + r) * HD;
    v16bf kf0, kf1;
    { BF16x16 u; u.u[0] = *(const uint4*)(krow + hi * 16);
      u.u[1] = *(const uint4*)(krow + hi * 16 + 8); kf0 = u.v; }
    { BF16x16 u; u.u[0] = *(const uint4*)(krow + 32 + hi * 16);
      u.u[1] = *(const uint4*)(krow + 32 + hi * 16 + 8); kf1 = u.v; }
    v8f c = v8f_zero();
    c = wmma_bf16(qf[0], kf0, c);
    c = wmma_bf16(qf[1], kf1, c);
    for (int i = 0; i < 8; ++i) {
      int q  = qbase + hi * 8 + i;
      int kk = n0 + r;
      float s = c[i] * 0.125f + ((kk <= q) ? 0.0f : -1e9f);
      float rm = s;
      for (int off = 1; off < 16; off <<= 1)
        rm = fmaxf(rm, __shfl_xor(rm, off, 32));
      float mn = fmaxf(m[i], rm);
      float e  = __expf(s - mn);
      float rs = e;
      for (int off = 1; off < 16; off <<= 1)
        rs += __shfl_xor(rs, off, 32);
      l[i] = l[i] * __expf(m[i] - mn) + rs;
      m[i] = mn;
    }
  }
  float linv[8];
  for (int i = 0; i < 8; ++i) linv[i] = 1.0f / l[i];

  // -------- pass 2: probs -> column sums + O = P @ V
  v8f o[4];
  for (int dc = 0; dc < 4; ++dc) o[dc] = v8f_zero();
  const int kend2 = (kend + 31) & ~31;

  for (int n0 = 0; n0 < kend2; n0 += 32) {
    float p0[8], p1[8];
    for (int t = 0; t < 2; ++t) {
      int nb = n0 + t * 16;
      const unsigned short* krow = Kb + (size_t)(nb + r) * HD;
      v16bf kf0, kf1;
      { BF16x16 u; u.u[0] = *(const uint4*)(krow + hi * 16);
        u.u[1] = *(const uint4*)(krow + hi * 16 + 8); kf0 = u.v; }
      { BF16x16 u; u.u[0] = *(const uint4*)(krow + 32 + hi * 16);
        u.u[1] = *(const uint4*)(krow + 32 + hi * 16 + 8); kf1 = u.v; }
      v8f c = v8f_zero();
      c = wmma_bf16(qf[0], kf0, c);
      c = wmma_bf16(qf[1], kf1, c);
      float* pp = t ? p1 : p0;
      for (int i = 0; i < 8; ++i) {
        int q  = qbase + hi * 8 + i;
        int kk = nb + r;
        float s = c[i] * 0.125f + ((kk <= q) ? 0.0f : -1e9f);
        pp[i] = __expf(s - m[i]) * linv[i];   // masked -> exp(-huge) == 0
      }
    }

    // column sums of probs (over the 16 query rows), one atomic per lane
    {
      float t0 = 0.0f, t1 = 0.0f;
      for (int i = 0; i < 8; ++i) { t0 += p0[i]; t1 += p1[i]; }
      t0 += __shfl_xor(t0, 16, 32);
      t1 += __shfl_xor(t1, 16, 32);
      if (hi == 0) atomicAdd(&ssum[n0 + r], t0);
      else         atomicAdd(&ssum[n0 + 16 + r], t1);
    }

    // transpose P (C layout) -> A-fragment layout through per-wave LDS
    for (int i = 0; i < 8; ++i) {
      pl[wave][hi * 8 + i][r]      = p0[i];
      pl[wave][hi * 8 + i][16 + r] = p1[i];
    }
    asm volatile("s_wait_dscnt 0x0" ::: "memory");  // wave-synchronous LDS
    BF16x16 u;
    for (int j = 0; j < 8; ++j) u.s[j]     = f2bfu(pl[wave][r][hi * 16 + j]);
    for (int j = 0; j < 8; ++j) u.s[8 + j] = f2bfu(pl[wave][r][hi * 16 + 8 + j]);
    v16bf pf = u.v;
    asm volatile("" ::: "memory");

    // V B-fragments: col = dc*16 + r, K(keys) = n0 + hi*16 + j
    for (int dc = 0; dc < 4; ++dc) {
      const unsigned short* vp = Vb + (size_t)n0 * HD + dc * 16 + r;
      BF16x16 uv;
      for (int j = 0; j < 16; ++j) uv.s[j] = vp[(size_t)(hi * 16 + j) * HD];
      o[dc] = wmma_bf16(pf, uv.v, o[dc]);
    }
  }

  // store O tile -> merged[q][h*64 + d]  (bf16)
  for (int dc = 0; dc < 4; ++dc)
    for (int i = 0; i < 8; ++i) {
      int q = qbase + hi * 8 + i;
      int d = dc * 16 + r;
      merged[(size_t)q * HDIM + h * HD + d] = f2bfu(o[dc][i]);
    }
}

// ------------------------------------------------------------ top-k mask
__global__ __launch_bounds__(256) void topk_kernel(const float* __restrict__ ssum,
                                                   float* __restrict__ mask) {
  __shared__ float vals[NSEL];
  __shared__ float redv[256];
  __shared__ int   redi[256];
  const int tid = threadIdx.x;
  for (int i = tid; i < NSEL; i += 256) vals[i] = ssum[i];
  for (int i = tid; i < S_LEN + 1; i += 256) mask[i] = (i <= NSEL) ? 0.0f : 1.0f;
  __syncthreads();
  for (int it = 0; it < HEAVY; ++it) {
    float best = -1e30f; int bi = 0;
    for (int i = tid; i < NSEL; i += 256)
      if (vals[i] > best) { best = vals[i]; bi = i; }
    redv[tid] = best; redi[tid] = bi;
    __syncthreads();
    for (int s2 = 128; s2 > 0; s2 >>= 1) {
      if (tid < s2 && redv[tid + s2] > redv[tid]) {
        redv[tid] = redv[tid + s2]; redi[tid] = redi[tid + s2];
      }
      __syncthreads();
    }
    if (tid == 0) { int w = redi[0]; mask[w] = 1.0f; vals[w] = -1e30f; }
    __syncthreads();
  }
}

// ---------------------------------------------------------------- driver
extern "C" void kernel_launch(void* const* d_in, const int* in_sizes, int n_in,
                              void* d_out, int out_size, void* d_ws, size_t ws_size,
                              hipStream_t stream) {
  (void)in_sizes; (void)n_in; (void)out_size; (void)ws_size;
  const float* hs      = (const float*)d_in[0];
  const float* w_qkv   = (const float*)d_in[1];
  const float* w_dense = (const float*)d_in[2];
  float* out      = (float*)d_out;
  float* mask_out = out + (size_t)S_LEN * HDIM;

  const size_t XB_BYTES = (size_t)S_LEN * HDIM * 2;     // also "merged" slot
  const size_t WB_BYTES = (size_t)HDIM * NQKV * 2;      // qkv then dense weights
  const size_t FU_BYTES = (size_t)S_LEN * NQKV * 4;
  const size_t QB_BYTES = (size_t)NHEADS * S_LEN * HD * 2;
  const size_t KB_BYTES = (size_t)S_LEN * HD * 2;

  char* ws = (char*)d_ws;
  unsigned short* Xb    = (unsigned short*)(ws);                       // reused as merged
  unsigned short* Wb    = (unsigned short*)(ws + XB_BYTES);
  float*          fused = (float*)(ws + XB_BYTES + WB_BYTES);
  unsigned short* Qb    = (unsigned short*)(ws + XB_BYTES + WB_BYTES + FU_BYTES);
  unsigned short* Kb    = (unsigned short*)(ws + XB_BYTES + WB_BYTES + FU_BYTES + QB_BYTES);
  unsigned short* Vb    = (unsigned short*)(ws + XB_BYTES + WB_BYTES + FU_BYTES + QB_BYTES + KB_BYTES);
  float*          ssum  = (float*)(ws + XB_BYTES + WB_BYTES + FU_BYTES + QB_BYTES + 2 * KB_BYTES);

  const int nX  = S_LEN * HDIM;
  const int nWq = HDIM * NQKV;
  const int nWd = HDIM * HDIM;

  // 1) bf16 converts
  f32_to_bf16_kernel<<<(nX + 255) / 256, 256, 0, stream>>>(hs, Xb, nX);
  f32_to_bf16_kernel<<<(nWq + 255) / 256, 256, 0, stream>>>(w_qkv, Wb, nWq);

  // 2) fused = X @ Wqkv
  gemm_bf16_kernel<NQKV, HDIM><<<dim3(NQKV / 64, S_LEN / 128), 256, 0, stream>>>(
      Xb, Wb, fused);

  // 3) dense weights into the same slot (qkv weights no longer needed)
  f32_to_bf16_kernel<<<(nWd + 255) / 256, 256, 0, stream>>>(w_dense, Wb, nWd);

  // 4) RoPE + split/convert
  {
    int total = S_LEN * 73 * HD;
    rope_kernel<<<(total + 255) / 256, 256, 0, stream>>>(fused, Qb, Kb, Vb);
  }

  // 5) attention (X slot reused as merged output)
  zero_f32_kernel<<<(S_LEN + 255) / 256, 256, 0, stream>>>(ssum, S_LEN);
  attn_kernel<<<dim3(S_LEN / 64, NHEADS), 128, 0, stream>>>(Qb, Kb, Vb, Xb, ssum);

  // 6) heavy-hitter mask
  topk_kernel<<<1, 256, 0, stream>>>(ssum, mask_out);

  // 7) out = merged @ Wdense
  gemm_bf16_kernel<HDIM, HDIM><<<dim3(HDIM / 64, S_LEN / 128), 256, 0, stream>>>(
      Xb, Wb, out);
}
